// NNUEEmbedding_74706661146752
// MI455X (gfx1250) — compile-verified, matrix operand order checked by reading, not run on hardware
//
#include <hip/hip_runtime.h>

// ---------------------------------------------------------------------------
// NNUE-style net on MI455X (gfx1250):
//   l0 = clip(concat(cur@W^T, oth@W^T) + b, 0, 1)   <-- HBM-bound big GEMM
//   l1 = clip(l0@W1^T + b1, 0, 1); l2 = clip(l1@W2^T + b2, 0, 1)
// Kernel 1: f16 WMMA GEMM, M=4096 N=256 K=41024, K split 16 ways,
//           f32 atomic reduction into d_ws accumulator (4 MiB).
// Kernel 2: bias + clip + tiny tail layers (f32 VALU).
// ---------------------------------------------------------------------------

typedef _Float16 v16h __attribute__((ext_vector_type(16)));
typedef _Float16 v8h  __attribute__((ext_vector_type(8)));
typedef float    v8f  __attribute__((ext_vector_type(8)));

#define B_ROWS  2048
#define M_TOT   4096
#define KDIM    41024
#define NDIM    256
#define M_TILE  128
#define KCHUNK  32
#define NCHUNKS (KDIM / KCHUNK)   // 1282
#define KSPLIT  16
#define WSTRIDE 40                 // halves per LDS row: 32 data + 8 pad (80 B)

__device__ __forceinline__ float clip01(float x) {
    return fminf(fmaxf(x, 0.0f), 1.0f);
}

__global__ __launch_bounds__(512)
void nnue_l0_gemm(const float* __restrict__ cur,
                  const float* __restrict__ oth,
                  const float* __restrict__ Wt,   // [256][41024] row-major
                  float* __restrict__ acc)        // [4096][256] f32, pre-zeroed
{
    __shared__ __align__(16) _Float16 Wlds[2][NDIM * WSTRIDE];   // 40 KiB

    const int tid    = threadIdx.x;
    const int lane   = tid & 31;
    const int wave   = tid >> 5;       // 0..15
    const int mstrip = wave & 7;       // 8 strips x 16 rows
    const int nstrip = wave >> 3;      // 2 strips x 128 cols

    const int mblock = blockIdx.x;     // 0..31
    const int ks     = blockIdx.y;     // 0..KSPLIT-1

    const float* Abase = (mblock < 16) ? cur : oth;
    const int row0 = (mblock & 15) * M_TILE;

    const int c0 = (NCHUNKS * ks) / KSPLIT;
    const int c1 = (NCHUNKS * (ks + 1)) / KSPLIT;

    // Per-lane A row pointer (16-bit A layout: lanes 16-31 carry k+8).
    const int am = row0 + mstrip * 16 + (lane & 15);
    const float* arow = Abase + (size_t)am * KDIM + ((lane >> 4) * 8);

    // W staging: 512 threads, 2 threads per N-row, 16 f32 (=16 halves) each.
    const int wn = tid >> 1;           // 0..255
    const int wh = tid & 1;            // which 16-wide half of the 32-k chunk
    const float* wrow = Wt + (size_t)wn * KDIM + wh * 16;

    v8f accr[8] = {};

    float4 a0, a1, a2, a3;             // A frag staging regs (16 f32)
    float4 q0, q1, q2, q3;             // W  stage regs      (16 f32)

    {   // prologue: load first chunk into regs
        const float4* ap = (const float4*)(arow + (size_t)c0 * KCHUNK);
        a0 = ap[0]; a1 = ap[1]; a2 = ap[4]; a3 = ap[5];
        const float4* wp = (const float4*)(wrow + (size_t)c0 * KCHUNK);
        q0 = wp[0]; q1 = wp[1]; q2 = wp[2]; q3 = wp[3];
    }

    for (int c = c0; c < c1; ++c) {
        const int buf = c & 1;

        // ---- stage W chunk (f32 -> f16) into LDS --------------------------
        _Float16* wd = &Wlds[buf][wn * WSTRIDE + wh * 16];
        v8h h0, h1;
        h0[0] = (_Float16)q0.x; h0[1] = (_Float16)q0.y;
        h0[2] = (_Float16)q0.z; h0[3] = (_Float16)q0.w;
        h0[4] = (_Float16)q1.x; h0[5] = (_Float16)q1.y;
        h0[6] = (_Float16)q1.z; h0[7] = (_Float16)q1.w;
        h1[0] = (_Float16)q2.x; h1[1] = (_Float16)q2.y;
        h1[2] = (_Float16)q2.z; h1[3] = (_Float16)q2.w;
        h1[4] = (_Float16)q3.x; h1[5] = (_Float16)q3.y;
        h1[6] = (_Float16)q3.z; h1[7] = (_Float16)q3.w;
        *(v8h*)wd       = h0;          // ds_store_b128
        *(v8h*)(wd + 8) = h1;          // ds_store_b128

        // ---- build A fragment (f32 -> f16) --------------------------------
        v16h af;
        af[0]  = (_Float16)a0.x; af[1]  = (_Float16)a0.y;
        af[2]  = (_Float16)a0.z; af[3]  = (_Float16)a0.w;
        af[4]  = (_Float16)a1.x; af[5]  = (_Float16)a1.y;
        af[6]  = (_Float16)a1.z; af[7]  = (_Float16)a1.w;
        af[8]  = (_Float16)a2.x; af[9]  = (_Float16)a2.y;
        af[10] = (_Float16)a2.z; af[11] = (_Float16)a2.w;
        af[12] = (_Float16)a3.x; af[13] = (_Float16)a3.y;
        af[14] = (_Float16)a3.z; af[15] = (_Float16)a3.w;

        // ---- issue next chunk's global loads (register double buffer) -----
        if (c + 1 < c1) {
            const float4* ap = (const float4*)(arow + (size_t)(c + 1) * KCHUNK);
            a0 = ap[0]; a1 = ap[1]; a2 = ap[4]; a3 = ap[5];
            const float4* wp = (const float4*)(wrow + (size_t)(c + 1) * KCHUNK);
            q0 = wp[0]; q1 = wp[1]; q2 = wp[2]; q3 = wp[3];
        }
        if (c + 8 < c1) {              // -> global_prefetch_b8
            __builtin_prefetch(arow + (size_t)(c + 8) * KCHUNK, 0, 0);
            __builtin_prefetch(wrow + (size_t)(c + 8) * KCHUNK, 0, 0);
        }

        __syncthreads();               // staged W visible to all waves

        // ---- 8 WMMAs: 16(M) x 128(N) x 32(K) ------------------------------
        const _Float16* wb = &Wlds[buf][0];
        const int koff = (lane >> 4) * 8;
#pragma unroll
        for (int t = 0; t < 8; ++t) {
            const int n = nstrip * 128 + t * 16 + (lane & 15);
            v8h lo = *(const v8h*)&wb[n * WSTRIDE + koff];        // ds_load_b128
            v8h hi = *(const v8h*)&wb[n * WSTRIDE + koff + 16];   // ds_load_b128
            v16h bf = __builtin_shufflevector(lo, hi,
                       0, 1, 2, 3, 4, 5, 6, 7, 8, 9, 10, 11, 12, 13, 14, 15);
            accr[t] = __builtin_amdgcn_wmma_f32_16x16x32_f16(
                          false, af, false, bf, (short)0, accr[t], false, false);
        }
        // single barrier per chunk is sufficient with double-buffered LDS
    }

    // ---- reduce partial tile into global accumulator ----------------------
    // C/D layout: vgpr v -> M=v (lanes 0-15), M=v+8 (lanes 16-31); N = lane%16
    const int mrow = mblock * M_TILE + mstrip * 16 + (lane >> 4) * 8;
    const int ncol = nstrip * 128 + (lane & 15);
#pragma unroll
    for (int t = 0; t < 8; ++t) {
#pragma unroll
        for (int v = 0; v < 8; ++v) {
            unsafeAtomicAdd(&acc[(size_t)(mrow + v) * NDIM + ncol + t * 16],
                            accr[t][v]);
        }
    }
}

__global__ __launch_bounds__(256)
void nnue_tail(const float* __restrict__ acc,   // [4096][256]
               const float* __restrict__ inb,   // [256]
               const float* __restrict__ w1,    // [32][512]
               const float* __restrict__ b1,    // [32]
               const float* __restrict__ w2,    // [32][32]
               const float* __restrict__ b2,    // [32]
               float* __restrict__ out)         // [2048][32]
{
    __shared__ float l0s[8][520];   // padded
    __shared__ float l1s[8][33];

    const int lane = threadIdx.x & 31;
    const int wave = threadIdx.x >> 5;
    const int r = blockIdx.x * 8 + wave;   // one row per wave

    // l0 = clip(concat(w_half, b_half) + bias)
#pragma unroll
    for (int j0 = 0; j0 < 256; j0 += 32) {
        const int j = j0 + lane;
        const float bias = inb[j];
        l0s[wave][j]       = clip01(acc[(size_t)r * NDIM + j] + bias);
        l0s[wave][256 + j] = clip01(acc[(size_t)(B_ROWS + r) * NDIM + j] + bias);
    }
    __syncthreads();

    // l1: lane j computes output neuron j over K=512 (LDS broadcast reads)
    float s = b1[lane];
    for (int k = 0; k < 512; ++k)
        s = fmaf(l0s[wave][k], w1[lane * 512 + k], s);
    l1s[wave][lane] = clip01(s);
    __syncthreads();

    // l2: lane j over K=32
    float s2 = b2[lane];
#pragma unroll
    for (int k = 0; k < 32; ++k)
        s2 = fmaf(l1s[wave][k], w2[lane * 32 + k], s2);
    out[(size_t)r * 32 + lane] = clip01(s2);
}

extern "C" void kernel_launch(void* const* d_in, const int* in_sizes, int n_in,
                              void* d_out, int out_size, void* d_ws, size_t ws_size,
                              hipStream_t stream)
{
    (void)in_sizes; (void)n_in; (void)out_size; (void)ws_size;

    const float* cur = (const float*)d_in[0];
    const float* oth = (const float*)d_in[1];
    const float* inW = (const float*)d_in[2];
    const float* inb = (const float*)d_in[3];
    const float* w1  = (const float*)d_in[4];
    const float* b1  = (const float*)d_in[5];
    const float* w2  = (const float*)d_in[6];
    const float* b2  = (const float*)d_in[7];
    float* out = (float*)d_out;
    float* acc = (float*)d_ws;   // needs M_TOT*NDIM*4 = 4 MiB of scratch

    hipMemsetAsync(acc, 0, (size_t)M_TOT * NDIM * sizeof(float), stream);
    nnue_l0_gemm<<<dim3(32, KSPLIT), 512, 0, stream>>>(cur, oth, inW, acc);
    nnue_tail<<<B_ROWS / 8, 256, 0, stream>>>(acc, inb, w1, b1, w2, b2, out);
}
